// SelfAttention_40123584479638
// MI455X (gfx1250) — matrix operand order repaired and checked by reading, not compiled, not run on hardware
//
#include <hip/hip_runtime.h>
#include <hip/hip_bf16.h>
#include <math.h>

#define N_EMBD   1024
#define N_HEAD   16
#define HEAD_DIM 64
#define BATCH    4
#define SEQ      2048
#define BT       (BATCH * SEQ) /* 8192 */

#define BK       32
#define NKSTEPS  (N_EMBD / BK)   /* 32 */
#define LTILE    (128 * BK)      /* halves per LDS tile buffer */

typedef _Float16 f16;
typedef __attribute__((ext_vector_type(16))) _Float16 v16h;
typedef __attribute__((ext_vector_type(8)))  _Float16 v8h;
typedef __attribute__((ext_vector_type(4)))  _Float16 v4h;
typedef __attribute__((ext_vector_type(8)))  float    v8f;

static __device__ __forceinline__ v8f wmma16(v16h a, v16h b, v8f c) {
    // D = A(16x32 f16) * B(32x16 f16) + C(16x16 f32)
    return __builtin_amdgcn_wmma_f32_16x16x32_f16(false, a, false, b, (short)0, c,
                                                  false, false);
}
static __device__ __forceinline__ v16h cat8(v8h a, v8h b) {
    return __builtin_shufflevector(a, b, 0,1,2,3,4,5,6,7,8,9,10,11,12,13,14,15);
}
static __device__ __forceinline__ v16h ldA(const f16* p) {   // A-frag: two 8-half chunks
    return cat8(*(const v8h*)p, *(const v8h*)(p + 8));
}

// ---------------------------------------------------------------- converts
__global__ void cvt_x_f16(const float* __restrict__ src, f16* __restrict__ dst, int n4) {
    int i = blockIdx.x * blockDim.x + threadIdx.x;
    int stride = gridDim.x * blockDim.x;
    for (; i < n4; i += stride) {
        float4 v = ((const float4*)src)[i];
        v4h h;
        h[0] = (f16)v.x; h[1] = (f16)v.y; h[2] = (f16)v.z; h[3] = (f16)v.w;
        ((v4h*)dst)[i] = h;
    }
}

// Wt[n][k] = (f16) W[k][n]   (both C x C)
__global__ void cvt_w_T(const float* __restrict__ W, f16* __restrict__ Wt) {
    int idx = blockIdx.x * blockDim.x + threadIdx.x;
    if (idx >= N_EMBD * N_EMBD) return;
    int n = idx >> 10;
    int k = idx & 1023;
    Wt[idx] = (f16)W[k * N_EMBD + n];
}

// ================================================================ GEMM core
// Block tile 128x128, BK=32, 8 waves (4M x 2N), wave tile 32x64.
// LDS double-buffered: global->LDS staging for step k+1 overlaps the WMMAs of
// step k; a single barrier per step (stores hit the buffer whose readers all
// passed the previous barrier). No register rotation => no accumulator moves.
#define GEMM_LDS_BODY(Asrc, Bsrc, acc)                                         \
    __shared__ f16 Al[2][LTILE];                                               \
    __shared__ f16 Bl[2][LTILE];                                               \
    const int tid   = threadIdx.x;                                             \
    const int lane  = tid & 31;                                                \
    const int wave  = tid >> 5;                                                \
    const int hi    = lane >> 4;                                               \
    const int ln    = lane & 15;                                               \
    const int waveM = wave & 3;                                                \
    const int waveN = wave >> 2;                                               \
    const int mblk  = blockIdx.x * 128;                                        \
    const int nblk  = blockIdx.y * 128;                                        \
    const int srow  = tid >> 1;                                                \
    const int scol  = (tid & 1) * 16;                                          \
    const f16* aG = (Asrc) + (mblk + srow) * N_EMBD + scol;                    \
    const f16* bG = (Bsrc) + (nblk + srow) * N_EMBD + scol;                    \
    f16* aL = &Al[0][srow * BK + scol];                                        \
    f16* bL = &Bl[0][srow * BK + scol];                                        \
    const f16* aF = &Al[0][(waveM * 32 + ln) * BK + hi * 16];                  \
    const f16* bF = &Bl[0][(waveN * 64 + ln) * BK + hi * 16];                  \
    {                                                                          \
        v16h a0 = *(const v16h*)(aG);                                          \
        v16h b0 = *(const v16h*)(bG);                                          \
        *(v16h*)aL = a0;                                                       \
        *(v16h*)bL = b0;                                                       \
    }                                                                          \
    __syncthreads();                                                           \
    for (int ks = 0; ks < NKSTEPS; ++ks) {                                     \
        const int  curo = (ks & 1) * LTILE;                                    \
        const bool more = (ks + 1) < NKSTEPS;                                  \
        v16h an, bn;                                                           \
        if (more) {                                                            \
            an = *(const v16h*)(aG + (ks + 1) * BK);                           \
            bn = *(const v16h*)(bG + (ks + 1) * BK);                           \
            if (ks + 2 < NKSTEPS) {                                            \
                __builtin_prefetch(aG + (ks + 2) * BK, 0, 1);                  \
                __builtin_prefetch(bG + (ks + 2) * BK, 0, 1);                  \
            }                                                                  \
        }                                                                      \
        const f16* ac = aF + curo;                                             \
        const f16* bc = bF + curo;                                             \
        v16h A0 = ldA(ac);                                                     \
        v16h A1 = ldA(ac + 16 * BK);                                           \
        v16h B0 = *(const v16h*)(bc);                                          \
        v16h B1 = *(const v16h*)(bc + 16 * BK);                                \
        v16h B2 = *(const v16h*)(bc + 32 * BK);                                \
        v16h B3 = *(const v16h*)(bc + 48 * BK);                                \
        acc[0][0] = wmma16(A0, B0, acc[0][0]);                                 \
        acc[1][0] = wmma16(A1, B0, acc[1][0]);                                 \
        acc[0][1] = wmma16(A0, B1, acc[0][1]);                                 \
        acc[1][1] = wmma16(A1, B1, acc[1][1]);                                 \
        acc[0][2] = wmma16(A0, B2, acc[0][2]);                                 \
        acc[1][2] = wmma16(A1, B2, acc[1][2]);                                 \
        acc[0][3] = wmma16(A0, B3, acc[0][3]);                                 \
        acc[1][3] = wmma16(A1, B3, acc[1][3]);                                 \
        if (more) {                                                            \
            const int nxto = LTILE - curo;                                     \
            *(v16h*)(aL + nxto) = an;                                          \
            *(v16h*)(bL + nxto) = bn;                                          \
            __syncthreads();                                                   \
        }                                                                      \
    }                                                                          \
    const int m0 = mblk + waveM * 32;                                          \
    const int n0 = nblk + waveN * 64;

// ---------------------------------------------------------------- QKV GEMM
// grid = (BT/128, C/128, 3), block = 256 (8 waves)
__global__ void __launch_bounds__(256)
qkv_gemm(const f16* __restrict__ xh,
         const f16* __restrict__ Wqt, const f16* __restrict__ Wkt, const f16* __restrict__ Wvt,
         const float* __restrict__ bq, const float* __restrict__ bk, const float* __restrict__ bv,
         f16* __restrict__ Q, f16* __restrict__ K, f16* __restrict__ V) {
    const f16*   Wt;
    const float* bias;
    f16*         dst;
    if (blockIdx.z == 0)      { Wt = Wqt; bias = bq; dst = Q; }
    else if (blockIdx.z == 1) { Wt = Wkt; bias = bk; dst = K; }
    else                      { Wt = Wvt; bias = bv; dst = V; }

    v8f acc[2][4] = {};
    GEMM_LDS_BODY(xh, Wt, acc)

    // store (bias fused) to [B, H, T, D] f16
#pragma unroll
    for (int nt = 0; nt < 4; ++nt) {
        int   col  = n0 + nt * 16 + ln;
        float bval = bias[col];
        int   h    = col >> 6;
        int   d    = col & 63;
#pragma unroll
        for (int mt = 0; mt < 2; ++mt) {
#pragma unroll
            for (int j = 0; j < 8; ++j) {
                int row = m0 + mt * 16 + j + hi * 8;    // global (b*T + t)
                int bb  = row >> 11;                    // /SEQ
                int t   = row & (SEQ - 1);
                dst[(((bb * N_HEAD + h) * SEQ) + t) * HEAD_DIM + d] =
                    (f16)(acc[mt][nt][j] + bval);
            }
        }
    }
}

// ---------------------------------------------------------------- out proj
// grid = (BT/128, C/128), block = 256
__global__ void __launch_bounds__(256)
out_proj(const f16* __restrict__ Ah, const f16* __restrict__ Wot,
         const float* __restrict__ bo, float* __restrict__ out) {
    v8f acc[2][4] = {};
    GEMM_LDS_BODY(Ah, Wot, acc)

#pragma unroll
    for (int nt = 0; nt < 4; ++nt) {
        int   col  = n0 + nt * 16 + ln;
        float bval = bo[col];
#pragma unroll
        for (int mt = 0; mt < 2; ++mt) {
#pragma unroll
            for (int j = 0; j < 8; ++j) {
                int row = m0 + mt * 16 + j + hi * 8;
                out[row * N_EMBD + col] = acc[mt][nt][j] + bval;
            }
        }
    }
}

// ---------------------------------------------------------------- attention
// grid = (T/64, B*H), block = 128 (4 waves). Wave handles 16 query rows.
// Double-buffered LDS staging of K / V^T tiles; one barrier per key tile.
__global__ void __launch_bounds__(128)
attn_fwd(const f16* __restrict__ Q, const f16* __restrict__ K,
         const f16* __restrict__ V, f16* __restrict__ Ah) {
    __shared__ f16 Kl[2][32 * 64];    // key tile, row-major [kk][d]
    __shared__ f16 Vt[2][64 * 32];    // value tile transposed [d][kk]
    __shared__ f16 Pl[4][16 * 32];    // per-wave P scratch

    const int lane = threadIdx.x & 31;
    const int wave = threadIdx.x >> 5;
    const int hi   = lane >> 4;
    const int ln   = lane & 15;
    const int bh   = blockIdx.y;
    const int b    = bh >> 4;
    const int h    = bh & 15;
    const int q0   = blockIdx.x * 64 + wave * 16;

    const f16* Qp = Q + (size_t)bh * SEQ * HEAD_DIM;
    const f16* Kp = K + (size_t)bh * SEQ * HEAD_DIM;
    const f16* Vp = V + (size_t)bh * SEQ * HEAD_DIM;

    // Q A-fragments, pre-scaled by 1/sqrt(D) = 2^-3 (exact in f16)
    const f16* qp  = Qp + (q0 + ln) * HEAD_DIM + hi * 16;
    const f16 qs = (f16)0.125f;
    v16h aq0 = ldA(qp)      * qs;
    v16h aq1 = ldA(qp + 32) * qs;

    v8f o0 = {}, o1 = {}, o2 = {}, o3 = {};
    float mrow[8], lrow[8];
#pragma unroll
    for (int j = 0; j < 8; ++j) { mrow[j] = -1e30f; lrow[j] = 0.0f; }

    const int qmax = q0 + 15;
    const int nkt  = blockIdx.x * 2 + 2;   // key tiles of 32 this block needs

    // staging indices (one v16h of K + one of V per thread per tile)
    const int flat = threadIdx.x * 16;     // 128 thr * 16 halves = 2048
    const int kr   = flat >> 6;
    const int d0   = flat & 63;

    // prologue: stage tile 0 into buffer 0
    {
        v16h kv = *(const v16h*)(Kp + kr * HEAD_DIM + d0);
        v16h vv = *(const v16h*)(Vp + kr * HEAD_DIM + d0);
        *(v16h*)(&Kl[0][kr * 64 + d0]) = kv;
#pragma unroll
        for (int i = 0; i < 16; ++i) Vt[0][(d0 + i) * 32 + kr] = vv[i];
    }
    __syncthreads();

    for (int kt = 0; kt < nkt; ++kt) {
        const int  kbase = kt * 32;
        const int  cur   = kt & 1;
        const bool more  = (kt + 1) < nkt;

        // issue global loads for next tile NOW; they complete under the WMMAs
        v16h kn, vn;
        if (more) {
            const f16* kp = Kp + (kbase + 32 + kr) * HEAD_DIM + d0;
            const f16* vp = Vp + (kbase + 32 + kr) * HEAD_DIM + d0;
            kn = *(const v16h*)kp;
            vn = *(const v16h*)vp;
        }

        if (kbase <= qmax) {   // wave-uniform: EXEC stays all-ones for WMMA
            const f16* klp = &Kl[cur][0];
            const f16* vtp = &Vt[cur][0];
            // S = Q*K^T, two 16-key column groups, K-dim 64 = 2 WMMAs each
            v8f s0 = {}, s1 = {};
            {
                v16h b0a = *(const v16h*)(klp + ln * 64 + hi * 16);
                v16h b0b = *(const v16h*)(klp + ln * 64 + 32 + hi * 16);
                v16h b1a = *(const v16h*)(klp + (16 + ln) * 64 + hi * 16);
                v16h b1b = *(const v16h*)(klp + (16 + ln) * 64 + 32 + hi * 16);
                s0 = wmma16(aq0, b0a, s0);
                s0 = wmma16(aq1, b0b, s0);
                s1 = wmma16(aq0, b1a, s1);
                s1 = wmma16(aq1, b1b, s1);
            }
            const int col0 = kbase + ln;
            const int col1 = kbase + 16 + ln;
#pragma unroll
            for (int j = 0; j < 8; ++j) {
                int   row = q0 + j + hi * 8;
                float x0  = s0[j]; if (col0 > row) x0 = -1e30f;
                float x1  = s1[j]; if (col1 > row) x1 = -1e30f;
                float rmax = fmaxf(x0, x1);
                rmax = fmaxf(rmax, __shfl_xor(rmax, 1, 32));
                rmax = fmaxf(rmax, __shfl_xor(rmax, 2, 32));
                rmax = fmaxf(rmax, __shfl_xor(rmax, 4, 32));
                rmax = fmaxf(rmax, __shfl_xor(rmax, 8, 32));
                float mn = fmaxf(mrow[j], rmax);
                float sc = __expf(mrow[j] - mn);
                float p0 = __expf(x0 - mn);
                float p1 = __expf(x1 - mn);
                float rs = p0 + p1;
                rs += __shfl_xor(rs, 1, 32);
                rs += __shfl_xor(rs, 2, 32);
                rs += __shfl_xor(rs, 4, 32);
                rs += __shfl_xor(rs, 8, 32);
                lrow[j] = lrow[j] * sc + rs;
                mrow[j] = mn;
                o0[j] *= sc; o1[j] *= sc; o2[j] *= sc; o3[j] *= sc;
                int prow = j + hi * 8;
                Pl[wave][prow * 32 + ln]      = (f16)p0;
                Pl[wave][prow * 32 + 16 + ln] = (f16)p1;
            }
            // P as A-fragment (per-wave LDS roundtrip; DS in-order per wave)
            const f16* pp = &Pl[wave][ln * 32 + hi * 16];
            v16h pa = cat8(*(const v8h*)pp, *(const v8h*)(pp + 8));
            // O += P * V  (4 d-column tiles)
            v16h bv0 = *(const v16h*)(vtp + (ln)      * 32 + hi * 16);
            v16h bv1 = *(const v16h*)(vtp + (16 + ln) * 32 + hi * 16);
            v16h bv2 = *(const v16h*)(vtp + (32 + ln) * 32 + hi * 16);
            v16h bv3 = *(const v16h*)(vtp + (48 + ln) * 32 + hi * 16);
            o0 = wmma16(pa, bv0, o0);
            o1 = wmma16(pa, bv1, o1);
            o2 = wmma16(pa, bv2, o2);
            o3 = wmma16(pa, bv3, o3);
        }

        if (more) {
            // stores target the *other* buffer; every wave already finished
            // reading it (barrier at end of iteration kt-1).
            const int nxt = 1 - cur;
            *(v16h*)(&Kl[nxt][kr * 64 + d0]) = kn;
#pragma unroll
            for (int i = 0; i < 16; ++i) Vt[nxt][(d0 + i) * 32 + kr] = vn[i];
            __syncthreads();
        }
    }

    // finalize: divide by row sums, write f16 [B, T, C] with C index = h*64+d
    float inv[8];
#pragma unroll
    for (int j = 0; j < 8; ++j) inv[j] = 1.0f / lrow[j];
    f16* outp = Ah + (size_t)b * SEQ * N_EMBD + h * HEAD_DIM;
#pragma unroll
    for (int j = 0; j < 8; ++j) {
        int t = q0 + j + hi * 8;
        f16* r = outp + t * N_EMBD + ln;
        r[0]  = (f16)(o0[j] * inv[j]);
        r[16] = (f16)(o1[j] * inv[j]);
        r[32] = (f16)(o2[j] * inv[j]);
        r[48] = (f16)(o3[j] * inv[j]);
    }
}

// ---------------------------------------------------------------- launch
extern "C" void kernel_launch(void* const* d_in, const int* in_sizes, int n_in,
                              void* d_out, int out_size, void* d_ws, size_t ws_size,
                              hipStream_t stream) {
    const float* x  = (const float*)d_in[0];
    const float* Wq = (const float*)d_in[1];
    const float* bq = (const float*)d_in[2];
    const float* Wk = (const float*)d_in[3];
    const float* bk = (const float*)d_in[4];
    const float* Wv = (const float*)d_in[5];
    const float* bv = (const float*)d_in[6];
    const float* Wo = (const float*)d_in[7];
    const float* bo = (const float*)d_in[8];
    float* out = (float*)d_out;

    char* ws = (char*)d_ws;
    const size_t SZ_X = (size_t)BT * N_EMBD * sizeof(f16);        // 16 MB
    const size_t SZ_W = (size_t)N_EMBD * N_EMBD * sizeof(f16);    // 2 MB
    f16* xh  = (f16*)ws;            ws += SZ_X;
    f16* Wqt = (f16*)ws;            ws += SZ_W;
    f16* Wkt = (f16*)ws;            ws += SZ_W;
    f16* Wvt = (f16*)ws;            ws += SZ_W;
    f16* Wot = (f16*)ws;            ws += SZ_W;
    f16* Qh  = (f16*)ws;            ws += SZ_X;   // [B,H,T,D]
    f16* Kh  = (f16*)ws;            ws += SZ_X;
    f16* Vh  = (f16*)ws;            ws += SZ_X;
    f16* Ah  = (f16*)ws;            ws += SZ_X;   // [B,T,C]

    cvt_x_f16<<<4096, 256, 0, stream>>>(x, xh, BT * N_EMBD / 4);
    cvt_w_T<<<(N_EMBD * N_EMBD + 255) / 256, 256, 0, stream>>>(Wq, Wqt);
    cvt_w_T<<<(N_EMBD * N_EMBD + 255) / 256, 256, 0, stream>>>(Wk, Wkt);
    cvt_w_T<<<(N_EMBD * N_EMBD + 255) / 256, 256, 0, stream>>>(Wv, Wvt);
    cvt_w_T<<<(N_EMBD * N_EMBD + 255) / 256, 256, 0, stream>>>(Wo, Wot);

    qkv_gemm<<<dim3(BT / 128, N_EMBD / 128, 3), 256, 0, stream>>>(
        xh, Wqt, Wkt, Wvt, bq, bk, bv, Qh, Kh, Vh);

    attn_fwd<<<dim3(SEQ / 64, BATCH * N_HEAD), 128, 0, stream>>>(Qh, Kh, Vh, Ah);

    out_proj<<<dim3(BT / 128, N_EMBD / 128), 256, 0, stream>>>(Ah, Wot, bo, out);
}
